// LSTMEncoder_807453851826
// MI455X (gfx1250) — matrix operand order repaired and checked by reading, not compiled, not run on hardware
//
#include <hip/hip_runtime.h>

#define B_    256
#define T_    512
#define D_    512
#define H_    512
#define FOURH 2048
#define KTOT  1024   // D + H, fused GEMM K dimension

#define BM  128      // block tile M (batch)
#define BN  128      // block tile N (interleaved gate units)
#define KB  32       // K chunk = one bf16 WMMA K
#define NK  (KTOT / KB)
#define LDA 40       // padded LDS K-stride in halfs (80B rows keep 16B align)
#define NBX 16
#define NBY 2
#define NBLOCKS (NBX * NBY)

// Use the Tensor Data Mover builtin only on the toolchain whose 5-arg
// signature is probe-verified (ROCm 7.2 ships no amd_gfx1250_TDM.h).
#if defined(__HIP_DEVICE_COMPILE__) && __has_builtin(__builtin_amdgcn_tensor_load_to_lds) && \
    !__has_include(<hip/amd_detail/amd_gfx1250_TDM.h>)
#define USE_TDM 1
#else
#define USE_TDM 0
#endif

typedef __attribute__((ext_vector_type(16))) __bf16 v16bf;
typedef __attribute__((ext_vector_type(8)))  __bf16 v8bf;
typedef __attribute__((ext_vector_type(8)))  float  v8f;
typedef __attribute__((ext_vector_type(4)))  unsigned int v4u;
typedef __attribute__((ext_vector_type(8)))  int v8i;
typedef __attribute__((ext_vector_type(4)))  int v4i;

__device__ __forceinline__ unsigned short f32_to_bf16(float f) {
    unsigned int u = __float_as_uint(f);
    u += 0x7FFFu + ((u >> 16) & 1u);          // round to nearest even
    return (unsigned short)(u >> 16);
}

__device__ __forceinline__ float sigmoidf_fast(float v) {
    return 1.0f / (1.0f + __expf(-v));
}

// ds_swizzle group-of-32: new_lane = lane ^ mask  (offset = mask<<10 | and 0x1f)
__device__ __forceinline__ float swz_xor1(float v) {
    return __int_as_float(__builtin_amdgcn_ds_swizzle(__float_as_int(v), 0x041F));
}
__device__ __forceinline__ float swz_xor2(float v) {
    return __int_as_float(__builtin_amdgcn_ds_swizzle(__float_as_int(v), 0x081F));
}

// Once per call: bf16, pre-transposed, GATE-INTERLEAVED weights
//   Wt[n'][k], n' = j*4 + gate; k: 0..511 Wx rows, 512..1023 Wh rows.
// Plus permuted bias, zeroed c / h0, zeroed grid-barrier words.
__global__ void lstm_init(const float* __restrict__ Wx,
                          const float* __restrict__ Wh,
                          const float* __restrict__ bias,
                          unsigned short* __restrict__ Wt,
                          float* __restrict__ biasPerm,
                          float* __restrict__ c,
                          float* __restrict__ h0,
                          unsigned* __restrict__ sync) {
    int tid = blockIdx.x * blockDim.x + threadIdx.x;   // 0 .. KTOT*FOURH-1
    int norig = tid & (FOURH - 1);
    int k = tid >> 11;                                  // 0..1023
    int g = norig >> 9;                                 // gate block i|f|g|o
    int j = norig & (H_ - 1);                           // hidden unit
    int nperm = (j << 2) | g;
    float w = (k < D_) ? Wx[(size_t)k * FOURH + norig]
                       : Wh[(size_t)(k - D_) * FOURH + norig];
    Wt[(size_t)nperm * KTOT + k] = f32_to_bf16(w);
    if (tid < FOURH) biasPerm[((tid & (H_ - 1)) << 2) | (tid >> 9)] = bias[tid];
    if (tid < B_ * H_) { c[tid] = 0.0f; h0[tid] = 0.0f; }
    if (tid < 2) sync[tid] = 0u;
}

__device__ __forceinline__ void grid_barrier(unsigned* cnt, unsigned* gen,
                                             unsigned target) {
    __threadfence();
    __syncthreads();
    if (threadIdx.x == 0) {
        unsigned arrived = atomicAdd(cnt, 1u);
        if (arrived == (unsigned)NBLOCKS - 1u) {
            atomicExch(cnt, 0u);
            __threadfence();
            atomicAdd(gen, 1u);
        } else {
            while (__hip_atomic_load(gen, __ATOMIC_RELAXED,
                                     __HIP_MEMORY_SCOPE_AGENT) < target)
                __builtin_amdgcn_s_sleep(2);
        }
    }
    __syncthreads();
}

// Persistent kernel: all 512 timesteps in one launch.
__global__ void __launch_bounds__(256)
lstm_persist(const float* __restrict__ x,              // [B,T,D] fp32
             const unsigned short* __restrict__ Wt,    // [4H][KTOT] bf16, K-major
             const float* __restrict__ biasPerm,       // [4H] permuted
             const int* __restrict__ lengths,          // [B]
             float* __restrict__ cS,                   // [B,H] cell state
             float* __restrict__ h0,                   // [B,H] ping
             float* __restrict__ h1,                   // [B,H] pong
             float* __restrict__ out,                  // [B,H]
             unsigned* __restrict__ sync) {
    __shared__ __align__(16) unsigned short Asm[2][BM * LDA];
    __shared__ __align__(16) unsigned short Bsm[2][BN * LDA];

    const int tid    = threadIdx.x;
    const int lane   = tid & 31;
    const int wave   = tid >> 5;
    const int waveM  = wave & 3;
    const int waveN  = wave >> 2;
    const int laneLo = lane & 15;
    const int laneHi = lane >> 4;

    const int M0 = blockIdx.y * BM;
    const int N0 = blockIdx.x * BN;

    const int sRow = tid & 127;
    const int sSeg = tid >> 7;

    unsigned* cnt = sync;
    unsigned* gen = sync + 1;

#if USE_TDM
    // D# group1 (constant): data_size=2B, pad 4 DWORDs after every 16 DWORDs,
    // tensor_dim0=KTOT, tensor_dim1=4H, tile 32x128, dim0 stride=KTOT.
    const v8i g1c = {0x06D10000, 0x04000000, 0x08000000, 0x00200000,
                     0x00000080, 0x00000400, 0, 0};
    const v4i gz = {0, 0, 0, 0};
#endif

#pragma unroll 1
    for (int t = 0; t < T_; ++t) {
        const float* hprev = (t & 1) ? h1 : h0;
        float* hnext = (t & 1) ? h0 : h1;

        v8f acc[2][4];
#pragma unroll
        for (int i = 0; i < 2; ++i)
#pragma unroll
            for (int jj = 0; jj < 4; ++jj) acc[i][jj] = {};

        auto stage = [&](int buf, int k0) {
            // ---- A: fp32 -> bf16 convert into LDS [M][K] ----
            {
                const int kh = sSeg * 16;
                const float* src = (k0 < D_)
                    ? x + (size_t)(M0 + sRow) * (T_ * D_) + (size_t)t * D_ + (k0 + kh)
                    : hprev + (size_t)(M0 + sRow) * H_ + (k0 - D_ + kh);
                float4 f0 = ((const float4*)src)[0];
                float4 f1 = ((const float4*)src)[1];
                float4 f2 = ((const float4*)src)[2];
                float4 f3 = ((const float4*)src)[3];
                uint4 p0, p1;
                p0.x = f32_to_bf16(f0.x) | ((unsigned)f32_to_bf16(f0.y) << 16);
                p0.y = f32_to_bf16(f0.z) | ((unsigned)f32_to_bf16(f0.w) << 16);
                p0.z = f32_to_bf16(f1.x) | ((unsigned)f32_to_bf16(f1.y) << 16);
                p0.w = f32_to_bf16(f1.z) | ((unsigned)f32_to_bf16(f1.w) << 16);
                p1.x = f32_to_bf16(f2.x) | ((unsigned)f32_to_bf16(f2.y) << 16);
                p1.y = f32_to_bf16(f2.z) | ((unsigned)f32_to_bf16(f2.w) << 16);
                p1.z = f32_to_bf16(f3.x) | ((unsigned)f32_to_bf16(f3.y) << 16);
                p1.w = f32_to_bf16(f3.z) | ((unsigned)f32_to_bf16(f3.w) << 16);
                uint4* dst = (uint4*)&Asm[buf][sRow * LDA + kh];
                dst[0] = p0;
                dst[1] = p1;
            }
            // ---- B: bf16 weights, TDM (uniform tile) or per-lane async ----
#if USE_TDM
            if (wave == 0) {
                unsigned long long ga =
                    (unsigned long long)(size_t)(Wt + (size_t)N0 * KTOT + k0);
                v4u g0 = {1u, (unsigned)(size_t)&Bsm[buf][0], (unsigned)ga,
                          (unsigned)((ga >> 32) & 0x01FFFFFFull) | 0x80000000u};
                __builtin_amdgcn_tensor_load_to_lds(g0, g1c, gz, gz, 0);
            }
#else
            {
                const unsigned short* srcb =
                    Wt + (size_t)(N0 + sRow) * KTOT + k0 + sSeg * 16;
                unsigned ldsb = (unsigned)(size_t)&Bsm[buf][sRow * LDA + sSeg * 16];
                unsigned long long ga = (unsigned long long)(size_t)srcb;
                asm volatile("global_load_async_to_lds_b128 %0, %1, off\n\t"
                             "global_load_async_to_lds_b128 %0, %1, off offset:16"
                             :: "v"(ldsb), "v"(ga) : "memory");
            }
#endif
        };

        auto waitB = [&]() {
#if USE_TDM
            __builtin_amdgcn_s_wait_tensorcnt(0);
#else
            asm volatile("s_wait_asynccnt 0" ::: "memory");
#endif
        };

        auto compute = [&](int buf) {
            v16bf aF[2];
#pragma unroll
            for (int ms = 0; ms < 2; ++ms) {
                int row = waveM * 32 + ms * 16 + laneLo;
                v8bf lo = *(const v8bf*)&Asm[buf][row * LDA + laneHi * 8];
                v8bf hi = *(const v8bf*)&Asm[buf][row * LDA + laneHi * 8 + 16];
                aF[ms] = __builtin_shufflevector(lo, hi, 0, 1, 2, 3, 4, 5, 6, 7,
                                                 8, 9, 10, 11, 12, 13, 14, 15);
            }
#pragma unroll
            for (int ns = 0; ns < 4; ++ns) {
                int col = waveN * 64 + ns * 16 + laneLo;
                v8bf blo = *(const v8bf*)&Bsm[buf][col * LDA + laneHi * 8];
                v8bf bhi = *(const v8bf*)&Bsm[buf][col * LDA + laneHi * 8 + 16];
                v16bf bF = __builtin_shufflevector(blo, bhi, 0, 1, 2, 3, 4, 5, 6,
                                                   7, 8, 9, 10, 11, 12, 13, 14, 15);
#pragma unroll
                for (int ms = 0; ms < 2; ++ms)
                    acc[ms][ns] = __builtin_amdgcn_wmma_f32_16x16x32_bf16(
                        false, aF[ms], false, bF, (short)0, acc[ms][ns], false, false);
            }
        };

        // ---- double-buffered pipeline: stage k+1 while WMMAs eat k ----
        stage(0, 0);
        waitB();
        __syncthreads();
        for (int it = 0; it < NK; ++it) {
            int cur = it & 1;
            if (it + 1 < NK) stage(cur ^ 1, (it + 1) * KB);
            compute(cur);
            waitB();
            __syncthreads();
        }

        // ---- fused epilogue: quad lanes hold gates i,f,g,o of one unit ----
#pragma unroll
        for (int ms = 0; ms < 2; ++ms) {
            int rowBase = M0 + waveM * 32 + ms * 16 + laneHi * 8;
#pragma unroll
            for (int ns = 0; ns < 4; ++ns) {
                int q = N0 + waveN * 64 + ns * 16 + laneLo;  // permuted column
                int j = q >> 2;
                int g = q & 3;
                float bb = biasPerm[q];
#pragma unroll
                for (int r = 0; r < 8; ++r) {
                    int row = rowBase + r;
                    float v  = acc[ms][ns][r] + bb;   // this lane: gate g
                    float x1 = swz_xor1(v);           // gate g^1
                    float x2 = swz_xor2(v);           // gate g^2
                    float x3 = swz_xor2(x1);          // gate g^3
                    float zi = (g == 0) ? v  : (g == 1) ? x1 : (g == 2) ? x2 : x3;
                    float zf = (g == 0) ? x1 : (g == 1) ? v  : (g == 2) ? x3 : x2;
                    float zg = (g == 0) ? x2 : (g == 1) ? x3 : (g == 2) ? v  : x1;
                    float zo = (g == 0) ? x3 : (g == 1) ? x2 : (g == 2) ? x1 : v;
                    float iv = sigmoidf_fast(zi);
                    float fv = sigmoidf_fast(zf);
                    float gv = tanhf(zg);
                    float ov = sigmoidf_fast(zo);
                    float cold = cS[(size_t)row * H_ + j];
                    float cnew = fv * cold + iv * gv;
                    float hnew = ov * tanhf(cnew);
                    if (g == 0) {
                        cS[(size_t)row * H_ + j] = cnew;
                        hnext[(size_t)row * H_ + j] = hnew;
                        int idx = lengths[row] - 1;
                        if (idx < 0) idx = 0;
                        if (t == idx) out[(size_t)row * H_ + j] = hnew;
                    }
                }
            }
        }

        grid_barrier(cnt, gen, (unsigned)(t + 1));
    }
}

extern "C" void kernel_launch(void* const* d_in, const int* in_sizes, int n_in,
                              void* d_out, int out_size, void* d_ws, size_t ws_size,
                              hipStream_t stream) {
    (void)in_sizes; (void)n_in; (void)out_size; (void)ws_size;
    const float* x    = (const float*)d_in[0];
    const int*   len  = (const int*)d_in[1];
    const float* Wx   = (const float*)d_in[2];
    const float* Wh   = (const float*)d_in[3];
    const float* bias = (const float*)d_in[4];
    float* out = (float*)d_out;

    char* ws = (char*)d_ws;
    unsigned short* Wt = (unsigned short*)ws;                   // 4 MiB bf16 weights
    float* c  = (float*)(ws + (4ull << 20));                    // 512 KiB cell state
    float* h0 = (float*)(ws + (4ull << 20) + (512ull << 10));   // 512 KiB h ping
    float* h1 = (float*)(ws + (5ull << 20));                    // 512 KiB h pong
    float* bp = (float*)(ws + (5ull << 20) + (512ull << 10));   // 8 KiB permuted bias
    unsigned* sync = (unsigned*)(ws + (5ull << 20) + (520ull << 10)); // barrier words

    lstm_init<<<(KTOT * FOURH) / 256, 256, 0, stream>>>(Wx, Wh, bias, Wt, bp, c, h0, sync);

    dim3 ggrid(NBX, NBY);   // 32 co-resident workgroups
    lstm_persist<<<ggrid, 256, 0, stream>>>(x, Wt, bp, len, c, h0, h1, out, sync);
}
